// CrossAttentionS2T_22204980921033
// MI455X (gfx1250) — matrix operand order repaired and checked by reading, not compiled, not run on hardware
//
#include <hip/hip_runtime.h>

// ---------------------------------------------------------------------------
// CrossAttentionS2T for MI455X (gfx1250, wave32, WMMA).
// All contractions use v_wmma_f32_16x16x32_bf16 (fp32 accumulate).
// Pipeline: prep(pos-add,bf16) -> GEMM Q -> GEMM KV (V stored transposed) ->
//           flash-attn (K/V staged to LDS via global_load_async_to_lds_b128,
//           shared by 8 waves) -> GEMM proj (scattered to output) -> CLS copy.
// ---------------------------------------------------------------------------

typedef __bf16 bf16;
typedef __attribute__((ext_vector_type(16))) __bf16 v16bf;
typedef __attribute__((ext_vector_type(8)))  float  v8f;

#define DIM   768
#define NH    12
#define HD    64
#define TT    8
#define NP    196
#define NB    4
#define SEQ   (NP * TT)      // 1568
#define ROWS  (NB * SEQ)     // 6272
#define QTILES (SEQ / 16)    // 98

__device__ __forceinline__ v8f wmma_bf16(v16bf a, v16bf b, v8f c) {
  // (neg_a, A, neg_b, B, c_mod, C, reuse_a, reuse_b)
  return __builtin_amdgcn_wmma_f32_16x16x32_bf16(false, a, false, b,
                                                 (short)0, c, false, false);
}

// A fragment: 16x32 bf16, row-major source, leading dim `ld`.
// Per lane this is one contiguous 32B run (k = 16*hb + s) -> 2x b128.
__device__ __forceinline__ v16bf load_A(const bf16* base, int ld) {
  int lane = threadIdx.x & 31;
  int m = lane & 15, hb = lane >> 4;
  const bf16* row = base + (size_t)m * ld;
  v16bf a;
#pragma unroll
  for (int v = 0; v < 4; ++v) {
    int k = 2 * v + 16 * hb;
    a[2 * v]     = row[k];
    a[2 * v + 1] = row[k + 1];
  }
#pragma unroll
  for (int v = 4; v < 8; ++v) {
    int k = 8 + 2 * (v - 4) + 16 * hb;
    a[2 * v]     = row[k];
    a[2 * v + 1] = row[k + 1];
  }
  return a;
}

// B fragment from a transposed source: B[k][n] = W[n][k], W row-major,
// ld = K-stride.  Per lane one contiguous 32B run -> 2x b128.
__device__ __forceinline__ v16bf load_BT(const bf16* w, int ld) {
  int lane = threadIdx.x & 31;
  int nn = lane & 15, hb = lane >> 4;
  const bf16* row = w + (size_t)nn * ld + 16 * hb;
  v16bf b;
#pragma unroll
  for (int v = 0; v < 8; ++v) {
    b[2 * v]     = row[2 * v];
    b[2 * v + 1] = row[2 * v + 1];
  }
  return b;
}

// Async global->LDS copy, 16B per active lane (ASYNCcnt-tracked).
__device__ __forceinline__ void async_copy_b128(const bf16* g, unsigned ldsoff) {
  asm volatile("global_load_async_to_lds_b128 %0, %1, off"
               :: "v"(ldsoff), "v"(g)
               : "memory");
}
__device__ __forceinline__ void wait_async0() {
  asm volatile("s_wait_asynccnt 0" ::: "memory");
}

// ---------------------------------------------------------------------------
__global__ void f2bf_kernel(const float* __restrict__ in, bf16* __restrict__ out,
                            int nelem) {
  int i = blockIdx.x * blockDim.x + threadIdx.x;
  if (i < nelem) out[i] = (bf16)in[i];
}

// pat[n, b*TT+t, :] + space_pos[n,:] + temp_pos[t,:] -> out[b, n*TT+t, :] bf16
__global__ void prep_kernel(const float* __restrict__ x,
                            const float* __restrict__ sp,
                            const float* __restrict__ tp,
                            int row_off, bf16* __restrict__ out) {
  size_t idx = (size_t)blockIdx.x * blockDim.x + threadIdx.x;
  if (idx >= (size_t)NB * NP * TT * DIM) return;
  int d = (int)(idx % DIM);
  size_t r = idx / DIM;
  int t = (int)(r % TT); r /= TT;
  int np = (int)(r % NP);
  int b  = (int)(r / NP);
  float v = x[((size_t)(np + row_off) * (NB * TT) + b * TT + t) * DIM + d]
          + sp[(size_t)np * DIM + d]
          + tp[(size_t)t * DIM + d];
  out[((size_t)b * SEQ + np * TT + t) * DIM + d] = (bf16)v;
}

// ---------------------------------------------------------------------------
// GEMM  C = X[rows x K] @ W[Ocols x K]^T + bias.  8 waves/block, each wave a
// 16x64 tile (4 accumulators); block covers 64 rows x 128 cols.
//   mode 0: Q    -> outq[(b,h,n,64)]  bf16 * 0.125 (SCALE folded, exact pow2)
//   mode 1: KV   -> outk[(b,h,m,64)] bf16; V TRANSPOSED: outv[(b,h,d),m] bf16
//   mode 2: PROJ -> f32 straight into d_out[(1+vp)*32 + b*8 + t][c]
// ---------------------------------------------------------------------------
__global__ __launch_bounds__(256) void gemm_bf16_kernel(
    const bf16* __restrict__ X, const bf16* __restrict__ W,
    const float* __restrict__ bias, int K, int mode,
    bf16* __restrict__ outq, bf16* __restrict__ outk,
    bf16* __restrict__ outv, float* __restrict__ outp) {
  int w    = threadIdx.x >> 5;
  int lane = threadIdx.x & 31;
  int nn = lane & 15, hb = lane >> 4;
  int row0 = blockIdx.y * 64 + (w & 3) * 16;
  int col0 = blockIdx.x * 128 + (w >> 2) * 64;

  v8f acc[4] = {};
  const bf16* xrow = X + (size_t)row0 * K;
  for (int kk = 0; kk < K; kk += 32) {
    __builtin_prefetch(xrow + kk + 64, 0, 1);   // global_prefetch_b8
    v16bf a = load_A(xrow + kk, K);
#pragma unroll
    for (int j = 0; j < 4; ++j) {
      v16bf bw = load_BT(W + (size_t)(col0 + 16 * j) * K + kk, K);
      acc[j] = wmma_bf16(a, bw, acc[j]);
    }
  }

#pragma unroll
  for (int j = 0; j < 4; ++j) {
    int c = col0 + 16 * j + nn;
    float bv = bias[c];
#pragma unroll
    for (int v = 0; v < 8; ++v) {
      int r = row0 + v + 8 * hb;
      float val = acc[j][v] + bv;
      int b_ = r / SEQ, ns = r % SEQ;
      if (mode == 0) {
        int h = c >> 6, d = c & 63;
        outq[(((size_t)(b_ * NH + h) * SEQ + ns) << 6) + d] = (bf16)(val * 0.125f);
      } else if (mode == 1) {
        int kv = c / DIM, c2 = c % DIM;
        int h = c2 >> 6, d = c2 & 63;
        if (kv == 0) {
          outk[(((size_t)(b_ * NH + h) * SEQ + ns) << 6) + d] = (bf16)val;
        } else {
          // transposed: [(b*NH+h)*HD + d][m], contiguous along m per lane
          outv[((size_t)(b_ * NH + h) * HD + d) * SEQ + ns] = (bf16)val;
        }
      } else {
        int vp = ns >> 3, t = ns & 7;
        outp[((size_t)(1 + vp) * 32 + b_ * TT + t) * DIM + c] = val;
      }
    }
  }
}

// ---------------------------------------------------------------------------
// Flash attention.  One block = 8 waves, all on the same (b,h); wave w owns
// query tile qt = blockIdx.x*8 + w (clamped; clamped waves skip final store).
// Per 32-key step the block async-copies K (32x64) and V^T (64x32) tiles into
// LDS once, shared by all 8 waves; fragments are read back as ds_load_b128.
// Online softmax in fp32 in the D-fragment layout; P transposed through
// per-wave LDS into the A-fragment layout for the PV WMMAs.
// ---------------------------------------------------------------------------
__global__ __launch_bounds__(256) void flash_attn_kernel(
    const bf16* __restrict__ Q, const bf16* __restrict__ Kb,
    const bf16* __restrict__ VT, bf16* __restrict__ AO) {
  __shared__ bf16 ldsK[32 * HD];      // [key 0..31][d 0..63]
  __shared__ bf16 ldsV[HD * 32];      // [d 0..63][key 0..31]
  __shared__ bf16 ldsP[8][16 * 32];   // per-wave P transpose scratch

  const int tid  = threadIdx.x;
  const int w    = tid >> 5;
  const int lane = tid & 31;
  const int nn = lane & 15, hb = lane >> 4;

  const int bh     = blockIdx.y;                 // b*NH + h
  const int qt_raw = blockIdx.x * 8 + w;
  const int qt     = qt_raw < QTILES ? qt_raw : QTILES - 1;

  const size_t bhs = (size_t)bh * SEQ;
  const bf16* qbase = Q + (bhs + (size_t)qt * 16) * HD;
  v16bf qa0 = load_A(qbase, HD);        // d = 0..31 (SCALE pre-folded into Q)
  v16bf qa1 = load_A(qbase + 32, HD);   // d = 32..63

  // Cooperative 16B-chunk addressing: 256 threads cover each 4KB tile.
  const bf16* kgsrc = Kb + (bhs + (tid >> 3)) * HD + (tid & 7) * 8;
  const bf16* vgsrc = VT + ((size_t)bh * HD + (tid >> 2)) * SEQ + (tid & 3) * 8;
  const unsigned kdst = (unsigned)(uintptr_t)&ldsK[0] + tid * 16;
  const unsigned vdst = (unsigned)(uintptr_t)&ldsV[0] + tid * 16;

  v8f o0 = {}, o1 = {}, o2 = {}, o3 = {};
  float mrow[8], lrow[8];
#pragma unroll
  for (int v = 0; v < 8; ++v) { mrow[v] = -3.0e38f; lrow[v] = 0.0f; }

  bf16* pt = ldsP[w];

  for (int kt = 0; kt < SEQ; kt += 32) {
    __syncthreads();                       // previous K/V tile fully consumed
    async_copy_b128(kgsrc + (size_t)kt * HD, kdst);
    async_copy_b128(vgsrc + kt, vdst);
    wait_async0();
    __syncthreads();                       // tile visible to all waves

    // S tile 16(queries) x 32(keys): two key halves, 2 K-steps over d each.
    v8f Sa = {}, Sb = {};
    Sa = wmma_bf16(qa0, load_BT(ldsK, HD), Sa);
    Sa = wmma_bf16(qa1, load_BT(ldsK + 32, HD), Sa);
    Sb = wmma_bf16(qa0, load_BT(ldsK + 16 * HD, HD), Sb);
    Sb = wmma_bf16(qa1, load_BT(ldsK + 16 * HD + 32, HD), Sb);

    // Online softmax (lane = key, row = vgpr + 8*half).
#pragma unroll
    for (int v = 0; v < 8; ++v) {
      float rm = fmaxf(Sa[v], Sb[v]);
#pragma unroll
      for (int m = 8; m >= 1; m >>= 1) rm = fmaxf(rm, __shfl_xor(rm, m, 32));
      float mnew  = fmaxf(mrow[v], rm);
      float scale = __expf(mrow[v] - mnew);
      float pa = __expf(Sa[v] - mnew);
      float pb = __expf(Sb[v] - mnew);
      float rs = pa + pb;
#pragma unroll
      for (int m = 8; m >= 1; m >>= 1) rs += __shfl_xor(rs, m, 32);
      lrow[v] = lrow[v] * scale + rs;
      mrow[v] = mnew;
      o0[v] *= scale; o1[v] *= scale; o2[v] *= scale; o3[v] *= scale;
      int r = v + 8 * hb;
      pt[r * 32 + nn]      = (bf16)pa;     // keys kt..kt+15
      pt[r * 32 + 16 + nn] = (bf16)pb;     // keys kt+16..kt+31
    }

    // P (16x32) back as A fragment; V^T tiles give contiguous B fragments.
    v16bf pfrag = load_A(pt, 32);
    o0 = wmma_bf16(pfrag, load_BT(ldsV + 0 * 16 * 32, 32), o0);
    o1 = wmma_bf16(pfrag, load_BT(ldsV + 1 * 16 * 32, 32), o1);
    o2 = wmma_bf16(pfrag, load_BT(ldsV + 2 * 16 * 32, 32), o2);
    o3 = wmma_bf16(pfrag, load_BT(ldsV + 3 * 16 * 32, 32), o3);
  }

  if (qt_raw < QTILES) {
    // Normalize, store bf16 reassembled to (b, n, h*64+d).
    int b = bh / NH, h = bh % NH;
    bf16* ao = AO + ((size_t)b * SEQ + (size_t)qt * 16) * DIM + h * HD;
#pragma unroll
    for (int v = 0; v < 8; ++v) {
      float inv = 1.0f / lrow[v];
      bf16* orow = ao + (size_t)(v + 8 * hb) * DIM;
      orow[nn]      = (bf16)(o0[v] * inv);
      orow[16 + nn] = (bf16)(o1[v] * inv);
      orow[32 + nn] = (bf16)(o2[v] * inv);
      orow[48 + nn] = (bf16)(o3[v] * inv);
    }
  }
}

// ---------------------------------------------------------------------------
__global__ void cls_copy_kernel(const float* __restrict__ t_x,
                                float* __restrict__ out) {
  int i = blockIdx.x * blockDim.x + threadIdx.x;
  if (i < 32 * DIM) out[i] = t_x[i];
}

// ---------------------------------------------------------------------------
extern "C" void kernel_launch(void* const* d_in, const int* in_sizes, int n_in,
                              void* d_out, int out_size, void* d_ws, size_t ws_size,
                              hipStream_t stream) {
  const float* s_x  = (const float*)d_in[0];
  const float* t_x  = (const float*)d_in[1];
  const float* csp  = (const float*)d_in[2];   // clip_space_pos
  const float* vsp  = (const float*)d_in[3];   // vmae_space_pos
  const float* ctp  = (const float*)d_in[4];   // clip_temporal_pos
  const float* vtp  = (const float*)d_in[5];   // vmae_temporal_pos
  const float* q_w  = (const float*)d_in[6];
  const float* q_b  = (const float*)d_in[7];
  const float* kv_w = (const float*)d_in[8];
  const float* kv_b = (const float*)d_in[9];
  const float* p_w  = (const float*)d_in[10];
  const float* p_b  = (const float*)d_in[11];
  float* out = (float*)d_out;

  char* ws = (char*)d_ws;
  size_t off = 0;
  auto alloc = [&](size_t bytes) -> void* {
    void* p = ws + off;
    off = (off + bytes + 255) & ~(size_t)255;
    return p;
  };
  bf16* t_pat  = (bf16*)alloc((size_t)ROWS * DIM * 2);
  bf16* s_pat  = (bf16*)alloc((size_t)ROWS * DIM * 2);
  bf16* qw_bf  = (bf16*)alloc((size_t)DIM * DIM * 2);
  bf16* kvw_bf = (bf16*)alloc((size_t)2 * DIM * DIM * 2);
  bf16* pw_bf  = (bf16*)alloc((size_t)DIM * DIM * 2);
  bf16* q_buf  = (bf16*)alloc((size_t)ROWS * DIM * 2);
  bf16* k_buf  = (bf16*)alloc((size_t)ROWS * DIM * 2);
  bf16* vT_buf = (bf16*)alloc((size_t)ROWS * DIM * 2);
  bf16* ao     = (bf16*)alloc((size_t)ROWS * DIM * 2);
  (void)ws_size; (void)in_sizes; (void)n_in; (void)out_size;

  // 1) weights -> bf16
  {
    int n1 = DIM * DIM, n2 = 2 * DIM * DIM;
    f2bf_kernel<<<(n1 + 255) / 256, 256, 0, stream>>>(q_w, qw_bf, n1);
    f2bf_kernel<<<(n2 + 255) / 256, 256, 0, stream>>>(kv_w, kvw_bf, n2);
    f2bf_kernel<<<(n1 + 255) / 256, 256, 0, stream>>>(p_w, pw_bf, n1);
  }

  // 2) positional prep (bf16 activations)
  {
    size_t n = (size_t)NB * NP * TT * DIM;
    int blocks = (int)((n + 255) / 256);
    prep_kernel<<<blocks, 256, 0, stream>>>(t_x, vsp, vtp, 1, t_pat);
    prep_kernel<<<blocks, 256, 0, stream>>>(s_x, csp, ctp, 2, s_pat);
  }

  // 3) Q = t_pat @ q_w^T + q_b   (scaled, per-head layout)
  gemm_bf16_kernel<<<dim3(DIM / 128, ROWS / 64), 256, 0, stream>>>(
      t_pat, qw_bf, q_b, DIM, 0, q_buf, nullptr, nullptr, nullptr);

  // 4) KV = s_pat @ kv_w^T + kv_b  (K per-head; V per-head TRANSPOSED)
  gemm_bf16_kernel<<<dim3(2 * DIM / 128, ROWS / 64), 256, 0, stream>>>(
      s_pat, kvw_bf, kv_b, DIM, 1, nullptr, k_buf, vT_buf, nullptr);

  // 5) attention -> ao (b, n, DIM) bf16
  flash_attn_kernel<<<dim3((QTILES + 7) / 8, NB * NH), 256, 0, stream>>>(
      q_buf, k_buf, vT_buf, ao);

  // 6) proj, scattered straight into output rows 1..196
  gemm_bf16_kernel<<<dim3(DIM / 128, ROWS / 64), 256, 0, stream>>>(
      ao, pw_bf, p_b, DIM, 2, nullptr, nullptr, nullptr, out);

  // 7) CLS row
  cls_copy_kernel<<<(32 * DIM + 255) / 256, 256, 0, stream>>>(t_x, out);
}
